// TransformerBlock_56530359550836
// MI455X (gfx1250) — compile-verified, matrix operand order checked by reading, not run on hardware
//
#include <hip/hip_runtime.h>

// ---------------------------------------------------------------------------
// MI455X / gfx1250. wave32, WMMA f32_16x16x32_f16, async global->LDS copies.
// Shapes: B=4, C=64, H=W=256, window/patch 8x8 (N=64), HIDDEN=170, C2=340.
// ---------------------------------------------------------------------------

typedef __attribute__((ext_vector_type(16))) _Float16 v16h;
typedef __attribute__((ext_vector_type(8)))  _Float16 v8h;
typedef __attribute__((ext_vector_type(8)))  float    v8f;
typedef __attribute__((ext_vector_type(4)))  float    v4f;
typedef __attribute__((ext_vector_type(4)))  int      v4i;

__device__ const float CT8[8] = {1.f, 0.70710678f, 0.f, -0.70710678f,
                                 -1.f, -0.70710678f, 0.f, 0.70710678f};
__device__ const float ST8[8] = {0.f, 0.70710678f, 1.f, 0.70710678f,
                                 0.f, -0.70710678f, -1.f, -0.70710678f};

// ---- async global -> LDS (gfx1250), with portable fallback ----------------
#ifdef __has_builtin
#if __has_builtin(__builtin_amdgcn_global_load_async_to_lds_b128) && \
    __has_builtin(__builtin_amdgcn_s_wait_asynccnt)
#define USE_ASYNC_LDS 1
#endif
#endif

typedef __attribute__((address_space(1))) v4i as1_v4i;  // global int4
typedef __attribute__((address_space(3))) v4i as3_v4i;  // LDS int4

__device__ __forceinline__ void async_copy16(const float* g, float* l) {
#if defined(USE_ASYNC_LDS)
  __builtin_amdgcn_global_load_async_to_lds_b128((as1_v4i*)g, (as3_v4i*)l, 0, 0);
#else
  *(v4f*)l = *(const v4f*)g;
#endif
}
__device__ __forceinline__ void async_wait_all() {
#if defined(USE_ASYNC_LDS)
  __builtin_amdgcn_s_wait_asynccnt(0);
#endif
}

// ---- WMMA fragment loaders ------------------------------------------------
// A/B fragment from a row-major half matrix in LDS (ld multiple of 8).
__device__ __forceinline__ v16h ld_frag_lds(const _Float16* m, int ld, int r0, int k0) {
  const int lane = threadIdx.x & 31;
  const _Float16* p = m + (r0 + (lane & 15)) * ld + k0 + ((lane >> 4) << 3);
  union { v16h v; v8h h[2]; } f;
  f.h[0] = *(const v8h*)p;        // ds_load_b128
  f.h[1] = *(const v8h*)(p + 16); // ds_load_b128
  return f.v;
}

// Fast unguarded B fragment from a row-major f32 matrix in global memory.
// Requires ld % 8 == 0 (32B-aligned rows) and all rows/K in range.
__device__ __forceinline__ v16h ld_frag_g32_fast(const float* m, int ld, int r0, int k0) {
  const int lane = threadIdx.x & 31;
  const float* p = m + (long)(r0 + (lane & 15)) * ld + k0 + ((lane >> 4) << 3);
  v4f a0 = *(const v4f*)p;
  v4f a1 = *(const v4f*)(p + 4);
  v4f b0 = *(const v4f*)(p + 16);
  v4f b1 = *(const v4f*)(p + 20);
  union { v16h v; _Float16 e[16]; } f;
#pragma unroll
  for (int j = 0; j < 4; ++j) {
    f.e[j]      = (_Float16)a0[j];
    f.e[4 + j]  = (_Float16)a1[j];
    f.e[8 + j]  = (_Float16)b0[j];
    f.e[12 + j] = (_Float16)b1[j];
  }
  return f.v;
}

// Guarded B fragment (row/K zero padding), for boundary tiles & odd ld.
__device__ __forceinline__ v16h ld_frag_g32(const float* m, int ld, int nrows,
                                            int kmax, int r0, int k0) {
  const int lane = threadIdx.x & 31;
  const int r  = r0 + (lane & 15);
  const int kb = k0 + ((lane >> 4) << 3);
  union { v16h v; _Float16 e[16]; } f;
#pragma unroll
  for (int j = 0; j < 16; ++j) f.e[j] = (_Float16)0.f;
  if (r < nrows) {
    const float* p = m + (long)r * ld;
#pragma unroll
    for (int j = 0; j < 8; ++j) {
      int k1 = kb + j, k2 = kb + 16 + j;
      if (k1 < kmax) f.e[j]     = (_Float16)p[k1];
      if (k2 < kmax) f.e[j + 8] = (_Float16)p[k2];
    }
  }
  return f.v;
}

#define WMMA_F16(A, Bm, Cm) \
  __builtin_amdgcn_wmma_f32_16x16x32_f16(false, (A), false, (Bm), (short)0, (Cm), false, false)

// ===========================================================================
// Kernel 1: LN1 + windowed attention + residual -> x1
// ===========================================================================
__global__ __launch_bounds__(256) void k1_attn(
    const float* __restrict__ x, const float* __restrict__ theta_max,
    const float* __restrict__ ln1w, const float* __restrict__ ln1b,
    const float* __restrict__ qkv_w, const float* __restrict__ qkv_b,
    const float* __restrict__ proj_w, const float* __restrict__ proj_b,
    const float* __restrict__ a_p, const float* __restrict__ b_p,
    const float* __restrict__ a_r, const float* __restrict__ b_r,
    float* __restrict__ x1) {
  __shared__ __align__(16) unsigned char smem[65536];
  float*    xs = (float*)smem;                 // [ch][tok] raw x (residual + LN in)
  float*    sB = (float*)(smem + 16384);       // [tok][tok] bias, then scores
  _Float16* an = (_Float16*)(smem + 32768);    // [tok][ch] LN out, later probs
  _Float16* qs = (_Float16*)(smem + 40960);    // [tok][ch] Q (scaled), later O
  _Float16* ks = (_Float16*)(smem + 49152);    // [tok][ch] K
  _Float16* vt = (_Float16*)(smem + 57344);    // [ch][tok] V transposed
  _Float16* ps = an;
  _Float16* os = qs;

  const int tid = threadIdx.x;
  const int wid = blockIdx.x;                  // 0..4095
  const int b   = wid >> 10;
  const int y0  = ((wid >> 5) & 31) << 3;
  const int x0  = (wid & 31) << 3;
  const float tmax = theta_max[b];

  if (tid < 48) __builtin_prefetch(qkv_w + tid * 256, 0, 0);
  if (tid < 16) __builtin_prefetch(proj_w + tid * 256, 0, 0);

  // async load window, transposed -> xs[ch][tok] (16B contiguous both sides)
  for (int i = tid; i < 1024; i += 256) {
    int c = i >> 4, ty = (i >> 1) & 7, xh = (i & 1) << 2;
    async_copy16(x + (((long)b * 64 + c) * 256 + y0 + ty) * 256 + x0 + xh,
                 xs + c * 64 + ty * 8 + xh);
  }
  // bias matrix (independent of xs, overlaps async copy latency)
  for (int i = tid; i < 4096; i += 256) {
    int n = i >> 6, m = i & 63;
    int az  = (n & 7) - (m & 7);
    int rad = (n >> 3) - (m >> 3);
    int azi = az < 0 ? az + 15 : az;
    int ri  = rad < 0 ? rad + 15 : rad;
    float aa = (float)az * (6.28318530717958647692f / 256.f);
    float rn = (float)rad * tmax * (1.f / 256.f);
    sB[i] = a_p[azi] * cosf(aa) + b_p[azi] * sinf(aa) +
            a_r[ri] * cosf(rn) + b_r[ri] * sinf(rn);
  }
  async_wait_all();
  __syncthreads();

  // LN1: 4 lanes per token, cross-lane reduce (wave32 shfl)
  {
    const int row = tid >> 2, part = tid & 3;
    float s = 0.f, s2 = 0.f;
#pragma unroll
    for (int j = 0; j < 16; ++j) {
      float v = xs[(part * 16 + j) * 64 + row];
      s += v; s2 += v * v;
    }
    s  += __shfl_xor(s, 1, 32);  s  += __shfl_xor(s, 2, 32);
    s2 += __shfl_xor(s2, 1, 32); s2 += __shfl_xor(s2, 2, 32);
    float mu  = s * (1.f / 64.f);
    float inv = rsqrtf(s2 * (1.f / 64.f) - mu * mu + 1e-5f);
#pragma unroll
    for (int j = 0; j < 16; ++j) {
      int c = part * 16 + j;
      an[row * 64 + c] = (_Float16)((xs[c * 64 + row] - mu) * inv * ln1w[c] + ln1b[c]);
    }
  }
  __syncthreads();

  const int lane = tid & 31;
  const int wvu  = __builtin_amdgcn_readfirstlane(tid >> 5);  // uniform wave id
  const int col  = lane & 15, rb = (lane >> 4) << 3;

  // Q (scaled by C^-0.5)
  for (int t = wvu; t < 16; t += 8) {
    int mt = t >> 2, nt = t & 3;
    v8f acc = {};
#pragma unroll
    for (int kk = 0; kk < 2; ++kk)
      acc = WMMA_F16(ld_frag_lds(an, 64, mt * 16, kk * 32),
                     ld_frag_g32_fast(qkv_w, 64, nt * 16, kk * 32), acc);
    float bv = qkv_b[nt * 16 + col];
#pragma unroll
    for (int r = 0; r < 8; ++r)
      qs[(mt * 16 + rb + r) * 64 + nt * 16 + col] = (_Float16)((acc[r] + bv) * 0.125f);
  }
  // K
  for (int t = wvu; t < 16; t += 8) {
    int mt = t >> 2, nt = t & 3;
    v8f acc = {};
#pragma unroll
    for (int kk = 0; kk < 2; ++kk)
      acc = WMMA_F16(ld_frag_lds(an, 64, mt * 16, kk * 32),
                     ld_frag_g32_fast(qkv_w, 64, 64 + nt * 16, kk * 32), acc);
    float bv = qkv_b[64 + nt * 16 + col];
#pragma unroll
    for (int r = 0; r < 8; ++r)
      ks[(mt * 16 + rb + r) * 64 + nt * 16 + col] = (_Float16)(acc[r] + bv);
  }
  // V (stored transposed [c][m])
  for (int t = wvu; t < 16; t += 8) {
    int mt = t >> 2, nt = t & 3;
    v8f acc = {};
#pragma unroll
    for (int kk = 0; kk < 2; ++kk)
      acc = WMMA_F16(ld_frag_lds(an, 64, mt * 16, kk * 32),
                     ld_frag_g32_fast(qkv_w, 64, 128 + nt * 16, kk * 32), acc);
    float bv = qkv_b[128 + nt * 16 + col];
#pragma unroll
    for (int r = 0; r < 8; ++r)
      vt[(nt * 16 + col) * 64 + mt * 16 + rb + r] = (_Float16)(acc[r] + bv);
  }
  __syncthreads();

  // S = Q K^T + bias
  for (int t = wvu; t < 16; t += 8) {
    int mt = t >> 2, nt = t & 3;
    v8f acc = {};
#pragma unroll
    for (int kk = 0; kk < 2; ++kk)
      acc = WMMA_F16(ld_frag_lds(qs, 64, mt * 16, kk * 32),
                     ld_frag_lds(ks, 64, nt * 16, kk * 32), acc);
#pragma unroll
    for (int r = 0; r < 8; ++r) {
      int idx = (mt * 16 + rb + r) * 64 + nt * 16 + col;
      sB[idx] = acc[r] + sB[idx];
    }
  }
  __syncthreads();

  // softmax over m, 4 lanes per row
  {
    const int row = tid >> 2, part = tid & 3;
    float mx = -1e30f;
#pragma unroll
    for (int j = 0; j < 16; ++j) mx = fmaxf(mx, sB[row * 64 + part * 16 + j]);
    mx = fmaxf(mx, __shfl_xor(mx, 1, 32));
    mx = fmaxf(mx, __shfl_xor(mx, 2, 32));
    float e[16], s = 0.f;
#pragma unroll
    for (int j = 0; j < 16; ++j) {
      e[j] = __expf(sB[row * 64 + part * 16 + j] - mx);
      s += e[j];
    }
    s += __shfl_xor(s, 1, 32); s += __shfl_xor(s, 2, 32);
    float invs = 1.f / s;
#pragma unroll
    for (int j = 0; j < 16; ++j)
      ps[row * 64 + part * 16 + j] = (_Float16)(e[j] * invs);
  }
  __syncthreads();

  // O = P V
  for (int t = wvu; t < 16; t += 8) {
    int mt = t >> 2, nt = t & 3;
    v8f acc = {};
#pragma unroll
    for (int kk = 0; kk < 2; ++kk)
      acc = WMMA_F16(ld_frag_lds(ps, 64, mt * 16, kk * 32),
                     ld_frag_lds(vt, 64, nt * 16, kk * 32), acc);
#pragma unroll
    for (int r = 0; r < 8; ++r)
      os[(mt * 16 + rb + r) * 64 + nt * 16 + col] = (_Float16)acc[r];
  }
  __syncthreads();

  // proj + residual -> x1
  for (int t = wvu; t < 16; t += 8) {
    int mt = t >> 2, nt = t & 3;
    v8f acc = {};
#pragma unroll
    for (int kk = 0; kk < 2; ++kk)
      acc = WMMA_F16(ld_frag_lds(os, 64, mt * 16, kk * 32),
                     ld_frag_g32_fast(proj_w, 64, nt * 16, kk * 32), acc);
    int oc = nt * 16 + col;
    float pb = proj_b[oc];
#pragma unroll
    for (int r = 0; r < 8; ++r) {
      int row = mt * 16 + rb + r;
      float v = acc[r] + pb + xs[oc * 64 + row];
      x1[(((long)b * 64 + oc) * 256 + y0 + (row >> 3)) * 256 + x0 + (row & 7)] = v;
    }
  }
}

// ===========================================================================
// Kernel 2: LN2 + project_in (64->340 WMMA) + gated 8x8 rFFT round-trip
// ===========================================================================
__global__ __launch_bounds__(256) void k2_fft(
    const float* __restrict__ x1, const float* __restrict__ ln2w,
    const float* __restrict__ ln2b, const float* __restrict__ pin_w,
    const float* __restrict__ gate, _Float16* __restrict__ hfft) {
  __shared__ __align__(16) unsigned char smem[53248];
  _Float16* hch = (_Float16*)smem;               // [352][64] post project_in
  float*    xs  = (float*)smem;                  // [ch][tok] overlays hch (phase-disjoint)
  _Float16* xn  = (_Float16*)(smem + 45056);     // [tok][ch] LN2 output

  const int tid = threadIdx.x;
  const int pid = blockIdx.x;
  const int b   = pid >> 10;
  const int y0  = ((pid >> 5) & 31) << 3;
  const int x0  = (pid & 31) << 3;

  if (tid < 85) __builtin_prefetch(pin_w + tid * 256, 0, 0);

  for (int i = tid; i < 1024; i += 256) {
    int c = i >> 4, ty = (i >> 1) & 7, xh = (i & 1) << 2;
    async_copy16(x1 + (((long)b * 64 + c) * 256 + y0 + ty) * 256 + x0 + xh,
                 xs + c * 64 + ty * 8 + xh);
  }
  async_wait_all();
  __syncthreads();

  {  // LN2, 4 lanes per pixel
    const int row = tid >> 2, part = tid & 3;
    float s = 0.f, s2 = 0.f;
#pragma unroll
    for (int j = 0; j < 16; ++j) {
      float v = xs[(part * 16 + j) * 64 + row];
      s += v; s2 += v * v;
    }
    s  += __shfl_xor(s, 1, 32);  s  += __shfl_xor(s, 2, 32);
    s2 += __shfl_xor(s2, 1, 32); s2 += __shfl_xor(s2, 2, 32);
    float mu  = s * (1.f / 64.f);
    float inv = rsqrtf(s2 * (1.f / 64.f) - mu * mu + 1e-5f);
#pragma unroll
    for (int j = 0; j < 16; ++j) {
      int c = part * 16 + j;
      xn[row * 64 + c] = (_Float16)((xs[c * 64 + row] - mu) * inv * ln2w[c] + ln2b[c]);
    }
  }
  __syncthreads();

  const int lane = tid & 31;
  const int wvu  = __builtin_amdgcn_readfirstlane(tid >> 5);
  const int col  = lane & 15, rb = (lane >> 4) << 3;

  // project_in: [64 pix x 64] @ [64 x 340] -> hch[ch][pix]  (N padded to 352)
  for (int t = wvu; t < 88; t += 8) {
    int mt = t & 3, nt = t >> 2;  // 4 M-tiles x 22 N-tiles
    v8f acc = {};
    if (nt < 21) {
#pragma unroll
      for (int kk = 0; kk < 2; ++kk)
        acc = WMMA_F16(ld_frag_lds(xn, 64, mt * 16, kk * 32),
                       ld_frag_g32_fast(pin_w, 64, nt * 16, kk * 32), acc);
    } else {
#pragma unroll
      for (int kk = 0; kk < 2; ++kk)
        acc = WMMA_F16(ld_frag_lds(xn, 64, mt * 16, kk * 32),
                       ld_frag_g32(pin_w, 64, 340, 64, nt * 16, kk * 32), acc);
    }
#pragma unroll
    for (int r = 0; r < 8; ++r)
      hch[(nt * 16 + col) * 64 + mt * 16 + rb + r] = (_Float16)acc[r];
  }
  __syncthreads();

  // per-channel gated FFT round trip (register DFTs, fully unrolled)
  for (int c = tid; c < 340; c += 256) {
    const _Float16* hp = hch + c * 64;
    float Xr[8][5], Xi[8][5];
#pragma unroll
    for (int y = 0; y < 8; ++y) {
#pragma unroll
      for (int v = 0; v < 5; ++v) {
        float sr = 0.f, si = 0.f;
#pragma unroll
        for (int xx = 0; xx < 8; ++xx) {
          float pv = (float)hp[y * 8 + xx];
          int ph = (v * xx) & 7;
          sr += pv * CT8[ph]; si -= pv * ST8[ph];
        }
        Xr[y][v] = sr; Xi[y][v] = si;
      }
    }
    float Gr[8][5], Gi[8][5];
#pragma unroll
    for (int u = 0; u < 8; ++u) {
#pragma unroll
      for (int v = 0; v < 5; ++v) {
        float fr = 0.f, fi = 0.f;
#pragma unroll
        for (int y = 0; y < 8; ++y) {
          int ph = (u * y) & 7;
          fr += Xr[y][v] * CT8[ph] + Xi[y][v] * ST8[ph];
          fi += Xi[y][v] * CT8[ph] - Xr[y][v] * ST8[ph];
        }
        float g = gate[c * 40 + u * 5 + v];
        Gr[u][v] = fr * g; Gi[u][v] = fi * g;
      }
    }
    float op[64];
#pragma unroll
    for (int i = 0; i < 64; ++i) op[i] = 0.f;
#pragma unroll
    for (int u = 0; u < 8; ++u) {
      const int mu = (8 - u) & 7;
#pragma unroll
      for (int xx = 0; xx < 8; ++xx) {
        float tr = 0.f, ti = 0.f;
#pragma unroll
        for (int v = 0; v < 5; ++v) {   // stored half-spectrum
          int ph = (v * xx) & 7;
          tr += Gr[u][v] * CT8[ph] - Gi[u][v] * ST8[ph];
          ti += Gr[u][v] * ST8[ph] + Gi[u][v] * CT8[ph];
        }
#pragma unroll
        for (int v = 5; v < 8; ++v) {   // Hermitian mirror
          int ph = (v * xx) & 7;
          float gr = Gr[mu][8 - v], gi = -Gi[mu][8 - v];
          tr += gr * CT8[ph] - gi * ST8[ph];
          ti += gr * ST8[ph] + gi * CT8[ph];
        }
#pragma unroll
        for (int y = 0; y < 8; ++y) {
          int ph = (u * y) & 7;
          op[y * 8 + xx] += tr * CT8[ph] - ti * ST8[ph];
        }
      }
    }
    long basep = ((long)b * 340 + c) * 65536;
#pragma unroll
    for (int p2 = 0; p2 < 64; ++p2)
      hfft[basep + (y0 + (p2 >> 3)) * 256 + x0 + (p2 & 7)] =
          (_Float16)(op[p2] * (1.f / 64.f));
  }
}

// ===========================================================================
// Kernel 3: depthwise 3x3 + GELU-gate + project_out + residual
// ===========================================================================
__device__ __forceinline__ float conv3x3(const _Float16* __restrict__ hf, long base,
                                         const float* __restrict__ w9, int gy, int gx) {
  float s = 0.f;
#pragma unroll
  for (int ky = 0; ky < 3; ++ky) {
    int yy = gy + ky - 1;
    if (yy < 0 || yy > 255) continue;
#pragma unroll
    for (int kx = 0; kx < 3; ++kx) {
      int xx = gx + kx - 1;
      if (xx < 0 || xx > 255) continue;
      s += (float)hf[base + (long)yy * 256 + xx] * w9[ky * 3 + kx];
    }
  }
  return s;
}

__global__ __launch_bounds__(256) void k3_out(
    const _Float16* __restrict__ hfft, const float* __restrict__ dw_w,
    const float* __restrict__ pout_w, const float* __restrict__ x1,
    float* __restrict__ out) {
  __shared__ __align__(16) _Float16 gbuf[64 * 192];  // [pix][hidden padded to 192]
  const int tid = threadIdx.x;
  const int pid = blockIdx.x;
  const int b   = pid >> 10;
  const int y0  = ((pid >> 5) & 31) << 3;
  const int x0  = (pid & 31) << 3;

  if (tid < 43) __builtin_prefetch(pout_w + tid * 256, 0, 0);

  for (int i = tid; i < 64 * 22; i += 256)
    gbuf[(i / 22) * 192 + 170 + (i % 22)] = (_Float16)0.f;   // K pad
  for (int i = tid; i < 170 * 64; i += 256) {
    int c = i >> 6, p = i & 63;
    int gy = y0 + (p >> 3), gx = x0 + (p & 7);
    long base1 = ((long)b * 340 + c) * 65536;
    long base2 = ((long)b * 340 + c + 170) * 65536;
    float c1 = conv3x3(hfft, base1, dw_w + c * 9, gy, gx);
    float c2 = conv3x3(hfft, base2, dw_w + (c + 170) * 9, gy, gx);
    float gl = 0.5f * c1 * (1.f + erff(c1 * 0.70710678118654752f));  // exact GELU
    gbuf[p * 192 + c] = (_Float16)(gl * c2);
  }
  __syncthreads();

  const int lane = tid & 31;
  const int wvu  = __builtin_amdgcn_readfirstlane(tid >> 5);
  const int col  = lane & 15, rb = (lane >> 4) << 3;
  for (int t = wvu; t < 16; t += 8) {
    int mt = t >> 2, nt = t & 3;
    v8f acc = {};
#pragma unroll
    for (int kk = 0; kk < 6; ++kk)
      acc = WMMA_F16(ld_frag_lds(gbuf, 192, mt * 16, kk * 32),
                     ld_frag_g32(pout_w, 170, 64, 170, nt * 16, kk * 32), acc);
    int oc = nt * 16 + col;
#pragma unroll
    for (int r = 0; r < 8; ++r) {
      int row = mt * 16 + rb + r;
      long gi = (((long)b * 64 + oc) * 256 + y0 + (row >> 3)) * 256 + x0 + (row & 7);
      out[gi] = acc[r] + x1[gi];
    }
  }
}

// ===========================================================================
extern "C" void kernel_launch(void* const* d_in, const int* in_sizes, int n_in,
                              void* d_out, int out_size, void* d_ws, size_t ws_size,
                              hipStream_t stream) {
  (void)in_sizes; (void)n_in; (void)out_size; (void)ws_size;
  const float* x      = (const float*)d_in[0];
  const float* theta  = (const float*)d_in[1];
  const float* ln1w   = (const float*)d_in[2];
  const float* ln1b   = (const float*)d_in[3];
  const float* qkv_w  = (const float*)d_in[4];
  const float* qkv_b  = (const float*)d_in[5];
  const float* proj_w = (const float*)d_in[6];
  const float* proj_b = (const float*)d_in[7];
  const float* a_p    = (const float*)d_in[8];
  const float* b_p    = (const float*)d_in[9];
  const float* a_r    = (const float*)d_in[10];
  const float* b_r    = (const float*)d_in[11];
  const float* ln2w   = (const float*)d_in[12];
  const float* ln2b   = (const float*)d_in[13];
  const float* pin_w  = (const float*)d_in[14];
  const float* dw_w   = (const float*)d_in[15];
  const float* fftg   = (const float*)d_in[16];
  const float* pout_w = (const float*)d_in[17];

  float*    x1   = (float*)d_ws;                                    // 64 MiB f32
  _Float16* hfft = (_Float16*)((char*)d_ws + (size_t)16777216 * 4); // 170 MiB f16
  float*    out  = (float*)d_out;

  dim3 blk(256);
  k1_attn<<<4096, blk, 0, stream>>>(x, theta, ln1w, ln1b, qkv_w, qkv_b,
                                    proj_w, proj_b, a_p, b_p, a_r, b_r, x1);
  k2_fft<<<4096, blk, 0, stream>>>(x1, ln2w, ln2b, pin_w, fftg, hfft);
  k3_out<<<4096, blk, 0, stream>>>(hfft, dw_w, pout_w, x1, out);
}
